// CrossTransLayer_66022237274257
// MI455X (gfx1250) — matrix operand order repaired and checked by reading, not compiled, not run on hardware
//
#include <hip/hip_runtime.h>
#include <hip/hip_bf16.h>
#include <stdint.h>

// ---------------- CDNA5 WMMA types ----------------
typedef __attribute__((ext_vector_type(16))) __bf16         v16bf;
typedef __attribute__((ext_vector_type(8)))  float          v8f;
typedef __attribute__((ext_vector_type(2)))  float          v2f;
typedef __attribute__((ext_vector_type(16))) unsigned short ushort16;
typedef __attribute__((ext_vector_type(8)))  unsigned short ushort8;
typedef __attribute__((ext_vector_type(4)))  unsigned short u16x4;  // NB: "ushort4" is taken by HIP headers

#if __has_builtin(__builtin_amdgcn_wmma_f32_16x16x4_f32)
#define HAVE_WMMA_F32 1
#else
#define HAVE_WMMA_F32 0
#endif

// ---------------- problem constants ----------------
static constexpr int cB  = 4;
static constexpr int cN  = 8192;
static constexpr int cNZ = 4096;
static constexpr int cD  = 512;     // DIM = H*DH
static constexpr int cH  = 8;
static constexpr int cDH = 64;
static constexpr int cM  = 256;     // landmarks
static constexpr int cG  = cB * cH; // 32 batched heads

// ---------------- helpers ----------------
// Hardware f32 -> bf16 convert (RNE); lowers to v_cvt_pk_bf16_f32 forms.
__device__ __forceinline__ unsigned short f2bf(float f) {
  __bf16 b = (__bf16)f;
  return __builtin_bit_cast(unsigned short, b);
}

__device__ __forceinline__ v8f wmma_bf16(v16bf a, v16bf b, v8f c) {
  return __builtin_amdgcn_wmma_f32_16x16x32_bf16(false, a, false, b, (short)0, c, false, false);
}
#if HAVE_WMMA_F32
__device__ __forceinline__ v8f wmma_f32(v2f a, v2f b, v8f c) {
  return __builtin_amdgcn_wmma_f32_16x16x4_f32(false, a, false, b, (short)0, c, false, false);
}
#endif

// A-matrix 16x32 bf16 frag: lanes 0-15 hold K {0..15}, lanes 16-31 {16..31}.
// rowp -> 32 contiguous bf16 K-values for this lane's row; two aligned 16B vector loads.
struct U16x16 { ushort8 lo, hi; };
__device__ __forceinline__ v16bf load_afrag(const unsigned short* rowp, int kh) {
  U16x16 t;
  t.lo = *(const ushort8*)(rowp + kh * 16);
  t.hi = *(const ushort8*)(rowp + kh * 16 + 8);
  return __builtin_bit_cast(v16bf, t);
}
// B frag where the 16 K-values are contiguous in memory (K-major operand): single 32B load
__device__ __forceinline__ v16bf load_bf16_contig(const unsigned short* p) {
  return *(const v16bf*)p;
}

// ---------------- CDNA5 async global->LDS copies (ASYNCcnt) ----------------
__device__ __forceinline__ void async_ld_b128(void* lds, const void* gaddr) {
  unsigned l = (unsigned)(uintptr_t)lds;   // low 32 bits of generic ptr = LDS byte offset
  asm volatile("global_load_async_to_lds_b128 %0, %1, off"
               :: "v"(l), "v"(gaddr) : "memory");
}
__device__ __forceinline__ void async_ld_b32(void* lds, const void* gaddr) {
  unsigned l = (unsigned)(uintptr_t)lds;
  asm volatile("global_load_async_to_lds_b32 %0, %1, off"
               :: "v"(l), "v"(gaddr) : "memory");
}
__device__ __forceinline__ void wait_async0() {
  asm volatile("s_wait_asynccnt 0" ::: "memory");
}

// ===================================================================
// Generic batched GEMM, f32 in/out, bf16 WMMA core, software-pipelined:
// next K-slab is register-staged while WMMA consumes the current slab
// from ping-pong LDS. B is stored K-contiguous per column so fragments
// are single ds_load_b128 pairs.
// C[M,N] = alpha * A[M,K] @ B(^T)[K,N]  (+bias[col]) (+res)
// batch g: offset = (g/IC)*sXo + (g%IC)*sXi
// Tiles: 64x64 per WG (8 waves), BK=32.  M,N mult of 64; K mult of 32.
// ===================================================================
__global__ __launch_bounds__(256) void k_gemm_bf16(
    const float* __restrict__ A, int lda, long sAo, long sAi,
    const float* __restrict__ B, int ldb, long sBo, long sBi, int transB,
    float* __restrict__ C, int ldc, long sCo, long sCi,
    int Kdim, int IC, float alpha,
    const float* __restrict__ bias, const float* __restrict__ res, int ldr)
{
  const int g = blockIdx.z, go = g / IC, gi = g % IC;
  A += go * sAo + gi * sAi;
  B += go * sBo + gi * sBi;
  C += go * sCo + gi * sCi;
  const int row0 = blockIdx.y * 64, col0 = blockIdx.x * 64;
  __shared__ alignas(32) unsigned short As[2][64 * 32];  // [row][k]
  __shared__ alignas(32) unsigned short Bs[2][64 * 32];  // [n][k]  (K-contiguous!)
  const int tid = threadIdx.x, w = tid >> 5, lane = tid & 31;
  const int rb = w & 3, cp = (w >> 2) * 2;       // wave's 16-row block, 2 col tiles
  v8f acc[2] = {{0,0,0,0,0,0,0,0},{0,0,0,0,0,0,0,0}};

  // per-thread staging coordinates (8 f32 each for A and B)
  const int ar = tid >> 2, ac = (tid & 3) * 8;   // A: 8 consecutive k of one row
  const int btn = tid >> 2, btc = (tid & 3) * 8; // B^T: 8 consecutive k of one col
  const int bk2 = (tid >> 4) * 2;                // B normal: 2 consecutive k ...
  const int bn4 = (tid & 15) * 4;                // ... x 4 consecutive n

  float aReg[8], bReg[8];
  auto loadAB = [&](int k0) {
    const float* pa = &A[(long)(row0 + ar) * lda + k0 + ac];
#pragma unroll
    for (int i = 0; i < 8; ++i) aReg[i] = pa[i];
    if (transB) {
      const float* pb = &B[(long)(col0 + btn) * ldb + k0 + btc];
#pragma unroll
      for (int i = 0; i < 8; ++i) bReg[i] = pb[i];
    } else {
      const float* pb0 = &B[(long)(k0 + bk2) * ldb + col0 + bn4];
      const float* pb1 = pb0 + ldb;
#pragma unroll
      for (int i = 0; i < 4; ++i) { bReg[i] = pb0[i]; bReg[4 + i] = pb1[i]; }
    }
  };
  auto storeS = [&](int p_) {
    unsigned short* Ap = As[p_];
    unsigned short* Bp = Bs[p_];
    ushort8 av;
#pragma unroll
    for (int i = 0; i < 8; ++i) av[i] = f2bf(aReg[i]);
    *(ushort8*)(&Ap[ar * 32 + ac]) = av;               // one 16B ds_store
    if (transB) {
      ushort8 bv;
#pragma unroll
      for (int i = 0; i < 8; ++i) bv[i] = f2bf(bReg[i]);
      *(ushort8*)(&Bp[btn * 32 + btc]) = bv;           // one 16B ds_store
    } else {
#pragma unroll
      for (int j = 0; j < 4; ++j) {                    // packed 2xbf16 dword stores
        const unsigned pk = (unsigned)f2bf(bReg[j]) | ((unsigned)f2bf(bReg[4 + j]) << 16);
        *(unsigned*)(&Bp[(bn4 + j) * 32 + bk2]) = pk;
      }
    }
  };

  const int nK = Kdim / 32;
  loadAB(0);
  for (int kt = 0; kt < nK; ++kt) {
    const int p_ = kt & 1;
    storeS(p_);
    __syncthreads();
    if (kt + 1 < nK) loadAB((kt + 1) * 32);   // overlap with WMMA below
    const int kh = lane >> 4;
    v16bf af = load_afrag(&As[p_][(rb * 16 + (lane & 15)) * 32], kh);
#pragma unroll
    for (int t = 0; t < 2; ++t) {
      v16bf bf = load_bf16_contig(&Bs[p_][((cp + t) * 16 + (lane & 15)) * 32 + kh * 16]);
      acc[t] = wmma_bf16(af, bf, acc[t]);
    }
    // no trailing barrier: next iteration writes the other buffer
  }

  const int lr = 8 * (lane >> 4), lc = lane & 15;
#pragma unroll
  for (int t = 0; t < 2; ++t) {
    const int gc = col0 + (cp + t) * 16 + lc;
    const float bv = bias ? bias[gc] : 0.0f;
#pragma unroll
    for (int r = 0; r < 8; ++r) {
      const long gr = row0 + rb * 16 + lr + r;
      float v = alpha * acc[t][r] + bv;
      if (res) v += res[gr * (long)ldr + gc];
      C[gr * (long)ldc + gc] = v;
    }
  }
}

// ===================================================================
// fp32 WMMA GEMM (V_WMMA_F32_16X16X4_F32) for the precision-sensitive
// Newton-Schulz chain. Tiles are raw f32 -> staged with CDNA5
// GLOBAL_LOAD_ASYNC_TO_LDS (ASYNCcnt) into ping-pong LDS buffers.
// ===================================================================
__global__ __launch_bounds__(256) void k_gemm_f32w(
    const float* __restrict__ A, int lda, long sAo, long sAi,
    const float* __restrict__ B, int ldb, long sBo, long sBi, int transB,
    float* __restrict__ C, int ldc, long sCo, long sCi,
    int Kdim, int IC, float alpha)
{
  const int g = blockIdx.z, go = g / IC, gi = g % IC;
  A += go * sAo + gi * sAi;
  B += go * sBo + gi * sBi;
  C += go * sCo + gi * sCi;
  const int row0 = blockIdx.y * 64, col0 = blockIdx.x * 64;
  __shared__ alignas(16) float Asf[2][64 * 32];  // [row][k]
  __shared__ alignas(16) float Bsf[2][32 * 64];  // [k][n]
  const int tid = threadIdx.x, w = tid >> 5, lane = tid & 31;
  const int rb = w & 3, cp = (w >> 2) * 2;
  v8f acc[2] = {{0,0,0,0,0,0,0,0},{0,0,0,0,0,0,0,0}};

  auto stage = [&](int k0, int p_) {
    // A tile 64x32: 2 x b128 per thread (16B aligned both sides)
#pragma unroll
    for (int j = 0; j < 2; ++j) {
      const int e = (tid + j * 256) * 4;
      const int r = e >> 5, c = e & 31;
      async_ld_b128(&Asf[p_][e], &A[(long)(row0 + r) * lda + k0 + c]);
    }
    if (!transB) {
      // B tile 32x64, global n-contiguous matches LDS [k][n]: 2 x b128
#pragma unroll
      for (int j = 0; j < 2; ++j) {
        const int e = (tid + j * 256) * 4;
        const int k = e >> 6, n = e & 63;
        async_ld_b128(&Bsf[p_][e], &B[(long)(k0 + k) * ldb + col0 + n]);
      }
    } else {
      // transposed B: element-wise async b32 copies
#pragma unroll
      for (int j = 0; j < 8; ++j) {
        const int e = tid + j * 256;
        const int k = e >> 6, n = e & 63;
        async_ld_b32(&Bsf[p_][e], &B[(long)(col0 + n) * ldb + k0 + k]);
      }
    }
  };

  const int nK = Kdim / 32;
  stage(0, 0);
  for (int kt = 0; kt < nK; ++kt) {
    const int p_ = kt & 1;
    wait_async0();        // own async copies into buf p_ complete
    __syncthreads();      // publish to workgroup
    if (kt + 1 < nK) stage((kt + 1) * 32, 1 - p_);  // DMA next slab during WMMA

    const int kh = lane >> 4;
    const int arow = rb * 16 + (lane & 15);
    const int bcol = cp * 16 + (lane & 15);
    const float* Ap = Asf[p_];
    const float* Bp = Bsf[p_];
#if HAVE_WMMA_F32
#pragma unroll
    for (int kk = 0; kk < 32; kk += 4) {
      // lanes 0-15: K=kk,kk+1 ; lanes 16-31: K=kk+2,kk+3  (one ds_load_b64)
      v2f a = *(const v2f*)(&Ap[arow * 32 + kk + 2 * kh]);
#pragma unroll
      for (int t = 0; t < 2; ++t) {
        v2f b;
        b[0] = Bp[(kk + 2 * kh) * 64 + bcol + t * 16];
        b[1] = Bp[(kk + 2 * kh + 1) * 64 + bcol + t * 16];
        acc[t] = wmma_f32(a, b, acc[t]);
      }
    }
#else
    ushort16 ua;
#pragma unroll
    for (int i = 0; i < 8; ++i) {
      ua[i]     = f2bf(Ap[arow * 32 + kh * 8 + i]);
      ua[8 + i] = f2bf(Ap[arow * 32 + 16 + kh * 8 + i]);
    }
    v16bf af = __builtin_bit_cast(v16bf, ua);
#pragma unroll
    for (int t = 0; t < 2; ++t) {
      ushort16 ub;
#pragma unroll
      for (int i = 0; i < 16; ++i) ub[i] = f2bf(Bp[(kh * 16 + i) * 64 + bcol + t * 16]);
      acc[t] = wmma_bf16(af, __builtin_bit_cast(v16bf, ub), acc[t]);
    }
#endif
  }
  const int lr = 8 * (lane >> 4), lc = lane & 15;
#pragma unroll
  for (int t = 0; t < 2; ++t) {
    const int gc = col0 + (cp + t) * 16 + lc;
#pragma unroll
    for (int r = 0; r < 8; ++r)
      C[(long)(row0 + rb * 16 + lr + r) * ldc + gc] = alpha * acc[t][r];
  }
}

// ===================================================================
// Landmark means: dst[g][m][d] = mean over grpLen consecutive rows
// ===================================================================
__global__ void k_landmarks(const float* __restrict__ src, int rowStride, int grpLen,
                            long batchStride, float* __restrict__ dst)
{
  const int m = blockIdx.x, g = blockIdx.y;
  const int b = g >> 3, h = g & 7;
  const float* p = src + (long)b * batchStride + (long)m * grpLen * rowStride + h * cDH;
  const int d = threadIdx.x;  // 64 threads
  float s = 0.0f;
  for (int t = 0; t < grpLen; ++t) s += p[(long)t * rowStride + d];
  dst[(((long)g * cM) + m) * cDH + d] = s / (float)grpLen;
}

// ===================================================================
// Row softmax in place, one block per row
// ===================================================================
__global__ __launch_bounds__(256) void k_softmax_rows(float* __restrict__ X, int cols, long ld)
{
  float* p = X + (long)blockIdx.x * ld;
  __shared__ float red[256];
  const int t = threadIdx.x;
  float m = -3.4e38f;
  for (int c = t; c < cols; c += 256) m = fmaxf(m, p[c]);
  red[t] = m; __syncthreads();
  for (int s = 128; s > 0; s >>= 1) { if (t < s) red[t] = fmaxf(red[t], red[t + s]); __syncthreads(); }
  m = red[0]; __syncthreads();
  float sum = 0.0f;
  for (int c = t; c < cols; c += 256) { float e = __expf(p[c] - m); p[c] = e; sum += e; }
  red[t] = sum; __syncthreads();
  for (int s = 128; s > 0; s >>= 1) { if (t < s) red[t] += red[t + s]; __syncthreads(); }
  const float inv = 1.0f / red[0];
  for (int c = t; c < cols; c += 256) p[c] *= inv;
}

// ===================================================================
// pinv init: |a| row sums (y==0) / col sums (y==1) for a[G][256][256]
// ===================================================================
__global__ __launch_bounds__(256) void k_abs_sums(const float* __restrict__ a,
                                                  float* __restrict__ rowsum,
                                                  float* __restrict__ colsum)
{
  const int idx = blockIdx.x, g = idx >> 8, r = idx & 255;
  __shared__ float red[256];
  const int t = threadIdx.x;
  const float* base = a + ((long)g << 16);
  red[t] = (blockIdx.y == 0) ? fabsf(base[r * 256 + t]) : fabsf(base[t * 256 + r]);
  __syncthreads();
  for (int s = 128; s > 0; s >>= 1) { if (t < s) red[t] += red[t + s]; __syncthreads(); }
  if (t == 0) (blockIdx.y == 0 ? rowsum : colsum)[idx] = red[0];
}

__global__ __launch_bounds__(256) void k_pinv_scale(const float* __restrict__ rowsum,
                                                    const float* __restrict__ colsum,
                                                    float* __restrict__ sOut, int n)
{
  __shared__ float red[256];
  const int t = threadIdx.x;
  float mr = 0.0f, mc = 0.0f;
  for (int i = t; i < n; i += 256) { mr = fmaxf(mr, rowsum[i]); mc = fmaxf(mc, colsum[i]); }
  red[t] = mr; __syncthreads();
  for (int s = 128; s > 0; s >>= 1) { if (t < s) red[t] = fmaxf(red[t], red[t + s]); __syncthreads(); }
  mr = red[0]; __syncthreads();
  red[t] = mc; __syncthreads();
  for (int s = 128; s > 0; s >>= 1) { if (t < s) red[t] = fmaxf(red[t], red[t + s]); __syncthreads(); }
  if (t == 0) sOut[0] = 1.0f / (mr * red[0]);
}

// zt0[g][i][j] = a[g][j][i] * s
__global__ void k_zt_init(const float* __restrict__ a, const float* __restrict__ s,
                          float* __restrict__ zt)
{
  const long i = (long)blockIdx.x * 256 + threadIdx.x;
  const long g = i >> 16; const int ij = (int)(i & 65535), r = ij >> 8, c = ij & 255;
  zt[i] = a[(g << 16) + ((long)c << 8) + r] * s[0];
}

// out = c*I - in  (elementwise, in-place safe)
__global__ void k_diag_sub(float* __restrict__ out, const float* __restrict__ in, float c)
{
  const long i = (long)blockIdx.x * 256 + threadIdx.x;
  const int ij = (int)(i & 65535), r = ij >> 8, cc = ij & 255;
  out[i] = ((r == cc) ? c : 0.0f) - in[i];
}

__global__ void k_conv_bf16(const float* __restrict__ in, unsigned short* __restrict__ out, long n)
{
  const long i = (long)blockIdx.x * 256 + threadIdx.x;
  if (i < n) out[i] = f2bf(in[i]);
}

// [G][256][64] f32 -> [G][64][256] bf16 (transpose so K is contiguous for B frags)
__global__ void k_conv_bf16_t(const float* __restrict__ in, unsigned short* __restrict__ out)
{
  const int g = blockIdx.x;
  const long base = (long)g * (cM * cDH);
  for (int i = threadIdx.x; i < cM * cDH; i += 256) {
    const int k = i >> 6, j = i & 63;
    out[base + (long)j * cM + k] = f2bf(in[base + i]);
  }
}

// ===================================================================
// Fused Nystrom output: for a 32-row q block,
//   out = softmax(q_blk @ k_land^T) @ W2      (never materializes attn1)
// ===================================================================
__global__ __launch_bounds__(256) void k_nystrom_out(
    const float* __restrict__ q,             // [B,N,512] (pre-scaled)
    const unsigned short* __restrict__ klb,  // [G,256,64] bf16
    const unsigned short* __restrict__ w2t,  // [G,64,256] bf16 (transposed W2)
    float* __restrict__ outp)                // [B,N,512]
{
  const int g = blockIdx.y, b = g >> 3, h = g & 7;
  const int row0 = blockIdx.x * 32;
  __shared__ alignas(32) float Sf[32 * 256];           // 32 KB
  __shared__ alignas(32) unsigned short Ps[32 * 256];  // 16 KB
  __shared__ alignas(32) unsigned short As[32 * 32];   //  2 KB
  __shared__ float part[8 * 32];
  __shared__ float rstat[32];
  const int tid = threadIdx.x, w = tid >> 5, lane = tid & 31;
  const float* qbh = q + ((long)b * cN + row0) * cD + h * cDH;
  const unsigned short* kl = klb + ((long)g << 14);
  const unsigned short* w2 = w2t + ((long)g << 14);

  // ---- S = q_blk @ k_land^T : [32 x 256]
  {
    const int rb = w & 1;
    const int c0 = (w >> 1) * 64;
    const int qr = tid >> 3, qc = (tid & 7) * 4;   // staging: 4 consecutive k per thread
    v8f acc[4] = {{0,0,0,0,0,0,0,0},{0,0,0,0,0,0,0,0},{0,0,0,0,0,0,0,0},{0,0,0,0,0,0,0,0}};
    for (int k0 = 0; k0 < cDH; k0 += 32) {
      {
        const float* ps = &qbh[(long)qr * cD + k0 + qc];
        u16x4 vq;
#pragma unroll
        for (int i = 0; i < 4; ++i) vq[i] = f2bf(ps[i]);
        *(u16x4*)(&As[qr * 32 + qc]) = vq;         // one 8B ds_store
      }
      __syncthreads();
      const int kh = lane >> 4;
      v16bf af = load_afrag(&As[(rb * 16 + (lane & 15)) * 32], kh);
#pragma unroll
      for (int t = 0; t < 4; ++t) {
        const int col = c0 + t * 16 + (lane & 15);
        v16bf bf = load_bf16_contig(&kl[(long)col * cDH + k0 + kh * 16]);
        acc[t] = wmma_bf16(af, bf, acc[t]);
      }
      __syncthreads();
    }
    const int lr = 8 * (lane >> 4), lc = lane & 15;
#pragma unroll
    for (int t = 0; t < 4; ++t)
#pragma unroll
      for (int r = 0; r < 8; ++r)
        Sf[(rb * 16 + lr + r) * 256 + c0 + t * 16 + lc] = acc[t][r];
  }
  __syncthreads();

  // ---- row softmax (32 rows x 256 cols), emit bf16 probs into Ps
  {
    const int r = tid & 31, seg = tid >> 5;
    float m = -3.4e38f;
    for (int c = seg * 32; c < seg * 32 + 32; ++c) m = fmaxf(m, Sf[r * 256 + c]);
    part[seg * 32 + r] = m; __syncthreads();
    if (tid < 32) {
      float mm = part[tid];
      for (int s = 1; s < 8; ++s) mm = fmaxf(mm, part[s * 32 + tid]);
      rstat[tid] = mm;
    }
    __syncthreads();
    const float rm = rstat[r];
    float sum = 0.0f;
    for (int c = seg * 32; c < seg * 32 + 32; ++c) {
      const float e = __expf(Sf[r * 256 + c] - rm);
      Sf[r * 256 + c] = e; sum += e;
    }
    part[seg * 32 + r] = sum; __syncthreads();
    if (tid < 32) {
      float ss = 0.0f;
      for (int s = 0; s < 8; ++s) ss += part[s * 32 + tid];
      rstat[tid] = 1.0f / ss;
    }
    __syncthreads();
    const float inv = rstat[r];
#pragma unroll
    for (int cc = 0; cc < 4; ++cc) {              // 16B vector stores of probs
      ushort8 pv;
#pragma unroll
      for (int i = 0; i < 8; ++i) pv[i] = f2bf(Sf[r * 256 + seg * 32 + cc * 8 + i] * inv);
      *(ushort8*)(&Ps[r * 256 + seg * 32 + cc * 8]) = pv;
    }
  }
  __syncthreads();

  // ---- O = P @ W2 : [32 x 64], one 16x16 tile per wave, K=256
  {
    const int rb = w & 1, cb = w >> 1;
    v8f acc = {0,0,0,0,0,0,0,0};
    const int kh = lane >> 4;
#pragma unroll
    for (int k0 = 0; k0 < 256; k0 += 32) {
      v16bf af = load_afrag(&Ps[(rb * 16 + (lane & 15)) * 256 + k0], kh);
      v16bf bf = load_bf16_contig(&w2[(long)(cb * 16 + (lane & 15)) * 256 + k0 + kh * 16]);
      acc = wmma_bf16(af, bf, acc);
    }
    const int lr = 8 * (lane >> 4), lc = lane & 15;
#pragma unroll
    for (int r = 0; r < 8; ++r) {
      const long gr = (long)b * cN + row0 + rb * 16 + lr + r;
      outp[gr * cD + h * cDH + cb * 16 + lc] = acc[r];
    }
  }
}

// ===================================================================
// Host orchestration
// ===================================================================
extern "C" void kernel_launch(void* const* d_in, const int* in_sizes, int n_in,
                              void* d_out, int out_size, void* d_ws, size_t ws_size,
                              hipStream_t stream)
{
  const float* x     = (const float*)d_in[0];
  const float* z     = (const float*)d_in[1];
  const float* W_q   = (const float*)d_in[2];
  const float* W_kv  = (const float*)d_in[3];
  const float* W_out = (const float*)d_in[4];
  const float* b_out = (const float*)d_in[5];
  float* out = (float*)d_out;
  (void)in_sizes; (void)n_in; (void)out_size; (void)ws_size;

  char* wp = (char*)d_ws;
  auto alloc = [&](size_t bytes) -> void* {
    void* p = (void*)wp;
    wp += (bytes + 255) & ~(size_t)255;
    return p;
  };
  float* q     = (float*)alloc((size_t)cB * cN * cD * 4);          // 64 MB
  float* kv    = (float*)alloc((size_t)cB * cNZ * 2 * cD * 4);     // 64 MB
  float* qland = (float*)alloc((size_t)cG * cM * cDH * 4);
  float* kland = (float*)alloc((size_t)cG * cM * cDH * 4);
  unsigned short* klb = (unsigned short*)alloc((size_t)cG * cM * cDH * 2);
  float* a2   = (float*)alloc((size_t)cG * cM * cM * 4);
  float* zt0  = (float*)alloc((size_t)cG * cM * cM * 4);
  float* zt1  = (float*)alloc((size_t)cG * cM * cM * 4);
  float* az   = (float*)alloc((size_t)cG * cM * cM * 4);
  float* tbuf = (float*)alloc((size_t)cG * cM * cM * 4);
  float* ubuf = (float*)alloc((size_t)cG * cM * cM * 4);
  float* rsum = (float*)alloc((size_t)cG * cM * 4);
  float* csum = (float*)alloc((size_t)cG * cM * 4);
  float* sscale = (float*)alloc(256);
  float* sim3 = (float*)alloc((size_t)cG * cM * cNZ * 4);          // 128 MB
  float* Tm   = (float*)alloc((size_t)cG * cM * cDH * 4);
  float* W2   = (float*)alloc((size_t)cG * cM * cDH * 4);
  unsigned short* w2t = (unsigned short*)alloc((size_t)cG * cDH * cM * 2);
  float* attn_out = sim3;  // sim3 is dead after T = attn3 @ v; reuse buffer

  const long szM = (long)cM * cM;   // 65536 per (b,h)
  const long szL = (long)cM * cDH;  // 16384 per (b,h)

  // 1. q = (x @ W_q) * DH^-0.5   [32768 x 512]
  k_gemm_bf16<<<dim3(cD / 64, (cB * cN) / 64, 1), 256, 0, stream>>>(
      x, cD, 0, 0, W_q, cD, 0, 0, 0,
      q, cD, 0, 0, cD, 1, 0.125f, nullptr, nullptr, 0);
  // 2. kv = z @ W_kv   [16384 x 1024]
  k_gemm_bf16<<<dim3((2 * cD) / 64, (cB * cNZ) / 64, 1), 256, 0, stream>>>(
      z, cD, 0, 0, W_kv, 2 * cD, 0, 0, 0,
      kv, 2 * cD, 0, 0, cD, 1, 1.0f, nullptr, nullptr, 0);
  // 3/4. landmark means
  k_landmarks<<<dim3(cM, cG), 64, 0, stream>>>(q, cD, cN / cM, (long)cN * cD, qland);
  k_landmarks<<<dim3(cM, cG), 64, 0, stream>>>(kv, 2 * cD, cNZ / cM, (long)cNZ * 2 * cD, kland);
  // 5. k_land -> bf16
  k_conv_bf16<<<dim3((cG * cM * cDH) / 256), 256, 0, stream>>>(kland, klb, (long)cG * cM * cDH);
  // 6. attn2 = q_land @ k_land^T (fp32 WMMA), then softmax
  k_gemm_f32w<<<dim3(cM / 64, cM / 64, cG), 256, 0, stream>>>(
      qland, cDH, 0, szL, kland, cDH, 0, szL, 1,
      a2, cM, 0, szM, cDH, cG, 1.0f);
  k_softmax_rows<<<dim3(cG * cM), 256, 0, stream>>>(a2, cM, cM);
  // 7. Newton-Schulz init: zt0 = a^T / (max(rowsum)*max(colsum))
  k_abs_sums<<<dim3(cG * cM, 2), 256, 0, stream>>>(a2, rsum, csum);
  k_pinv_scale<<<1, 256, 0, stream>>>(rsum, csum, sscale, cG * cM);
  k_zt_init<<<dim3((int)((cG * szM) / 256)), 256, 0, stream>>>(a2, sscale, zt0);

  auto gemm256 = [&](float* Cp, const float* Ap, const float* Bp, float alpha) {
    k_gemm_f32w<<<dim3(4, 4, cG), 256, 0, stream>>>(
        Ap, cM, 0, szM, Bp, cM, 0, szM, 0, Cp, cM, 0, szM, cM, cG, alpha);
  };
  auto diag = [&](float* o, const float* i, float c) {
    k_diag_sub<<<dim3((int)((cG * szM) / 256)), 256, 0, stream>>>(o, i, c);
  };
  // 8. zt = 0.25 * zt @ (13I - az@(15I - az@(7I - az))), 6 iterations
  for (int it = 0; it < 6; ++it) {
    float* zin  = (it & 1) ? zt1 : zt0;
    float* zout = (it & 1) ? zt0 : zt1;
    gemm256(az, a2, zin, 1.0f);
    diag(tbuf, az, 7.0f);
    gemm256(ubuf, az, tbuf, 1.0f);
    diag(ubuf, ubuf, 15.0f);
    gemm256(tbuf, az, ubuf, 1.0f);
    diag(tbuf, tbuf, 13.0f);
    gemm256(zout, zin, tbuf, 0.25f);
  }
  float* ztF = zt0;  // iteration 5 (odd) writes zt0

  // 9. sim3 = q_land @ k^T  [256 x 4096] per head, then softmax
  k_gemm_bf16<<<dim3(cNZ / 64, cM / 64, cG), 256, 0, stream>>>(
      qland, cDH, (long)cH * szL, szL,
      kv, 2 * cD, (long)cNZ * 2 * cD, cDH, 1,
      sim3, cNZ, (long)cH * cM * cNZ, (long)cM * cNZ,
      cDH, cH, 1.0f, nullptr, nullptr, 0);
  k_softmax_rows<<<dim3(cG * cM), 256, 0, stream>>>(sim3, cNZ, cNZ);
  // 10. T = attn3 @ v  [256 x 64] per head
  k_gemm_bf16<<<dim3(cDH / 64, cM / 64, cG), 256, 0, stream>>>(
      sim3, cNZ, (long)cH * cM * cNZ, (long)cM * cNZ,
      kv + cD, 2 * cD, (long)cNZ * 2 * cD, cDH, 0,
      Tm, cDH, (long)cH * szL, szL,
      cNZ, cH, 1.0f, nullptr, nullptr, 0);
  // 11. W2 = pinv(attn2) @ T  (fp32 WMMA)
  k_gemm_f32w<<<dim3(cDH / 64, cM / 64, cG), 256, 0, stream>>>(
      ztF, cM, 0, szM, Tm, cDH, 0, szL, 0,
      W2, cDH, 0, szL, cM, cG, 1.0f);
  k_conv_bf16_t<<<dim3(cG), 256, 0, stream>>>(W2, w2t);
  // 12. fused: attn_out = softmax(q @ k_land^T) @ W2
  k_nystrom_out<<<dim3(cN / 32, cG), 256, 0, stream>>>(q, klb, w2t, attn_out);
  // 13. out = x + attn_out @ W_out + b_out
  k_gemm_bf16<<<dim3(cD / 64, (cB * cN) / 64, 1), 256, 0, stream>>>(
      attn_out, cD, 0, 0, W_out, cD, 0, 0, 0,
      out, cD, 0, 0, cD, 1, 1.0f, b_out, x, cD);
}
